// Generator_41807211659736
// MI455X (gfx1250) — matrix operand order repaired and checked by reading, not compiled, not run on hardware
//
#include <hip/hip_runtime.h>
#include <hip/hip_bf16.h>

typedef __attribute__((ext_vector_type(16))) _Float16 v16h;
typedef __attribute__((ext_vector_type(8)))  _Float16 v8h;
typedef __attribute__((ext_vector_type(8)))  float    v8f;

#define B_TOT   4096
#define T_LEN   100
#define IN_DIM  8
#define HID     128
#define NCLS    10
#define STEPS   198
#define DT_C    0.05f

#define WAVES           4
#define BLOCK           (WAVES * 32)
#define ROWS_PER_WAVE   16
#define ROWS_PER_BLOCK  (WAVES * ROWS_PER_WAVE)

// LDS layout (offsets in _Float16 elements)
#define OFF_WF1T  0
#define OFF_WF2T  16384
#define OFF_WG1T  32768
#define OFF_WG2T  49152
#define OFF_WC1T  65536            // [16 N][128 K]
#define OFF_WC2T  67584            // [16 N][32 K] zero-padded
#define OFF_STAGE 68096            // per-wave [16 M][128 F] f16 staging
#define SMEM_HALVES (OFF_STAGE + WAVES * ROWS_PER_WAVE * HID)   // 76288 -> 152576 B

__device__ __forceinline__ float fast_tanh(float x) {
#if __has_builtin(__builtin_amdgcn_tanhf)
  return __builtin_amdgcn_tanhf(x);
#elif __has_builtin(__builtin_amdgcn_tanh_f32)
  return __builtin_amdgcn_tanh_f32(x);
#else
  float e = __expf(2.0f * x);
  return 1.0f - 2.0f / (e + 1.0f);
#endif
}

// Load a 16-half fragment as two 16B LDS reads (ds_load_b128 x2).
__device__ __forceinline__ v16h ld_frag(const _Float16* p, int second_off) {
  v8h lo = *(const v8h*)p;
  v8h hi = *(const v8h*)(p + second_off);
  return __builtin_shufflevector(lo, hi, 0,1,2,3,4,5,6,7,8,9,10,11,12,13,14,15);
}

__device__ __forceinline__ v8f wmma16(v16h a, v16h b, v8f c) {
  // D = A(16x32 f16) x B(32x16 f16) + C(16x16 f32)
  return __builtin_amdgcn_wmma_f32_16x16x32_f16(false, a, false, b, (short)0, c,
                                                false, false);
}

// One 16x128 @ 128x128 GEMM pass, software-pipelined:
//   - B fragments for N-tile nt+1 are issued BEFORE the epilogue of tile nt,
//     so ds_load latency overlaps with WMMA + TRANS epilogue work.
template <typename EPI>
__device__ __forceinline__ void gemm_pass(const _Float16* __restrict__ wt,
                                          const v16h a[4],
                                          const float bias[8],
                                          int c, int hi, EPI&& epi) {
  const _Float16* base = wt + c * HID + hi * 8;
  v16h b[4];
#pragma unroll
  for (int kt = 0; kt < 4; ++kt)
    b[kt] = ld_frag(base + kt * 32, 16);
#pragma unroll
  for (int nt = 0; nt < 8; ++nt) {
    v8f acc;
#pragma unroll
    for (int i = 0; i < 8; ++i) acc[i] = bias[nt];
#pragma unroll
    for (int kt = 0; kt < 4; ++kt)
      acc = wmma16(a[kt], b[kt], acc);
    v16h bn[4];
    if (nt < 7) {
      const _Float16* nb = base + (nt + 1) * (16 * HID);
#pragma unroll
      for (int kt = 0; kt < 4; ++kt)
        bn[kt] = ld_frag(nb + kt * 32, 16);
    }
    epi(nt, acc);
    if (nt < 7) {
#pragma unroll
      for (int kt = 0; kt < 4; ++kt) b[kt] = bn[kt];
    }
  }
}

__global__ void __launch_bounds__(BLOCK, 1) sde_fused_kernel(
    const float* __restrict__ coeffs, const float* __restrict__ dW,
    const float* __restrict__ Winit,  const float* __restrict__ binit,
    const float* __restrict__ Wf1, const float* __restrict__ bf1,
    const float* __restrict__ Wf2, const float* __restrict__ bf2,
    const float* __restrict__ Wg1, const float* __restrict__ bg1,
    const float* __restrict__ Wg2, const float* __restrict__ bg2,
    const float* __restrict__ Wc1, const float* __restrict__ bc1,
    const float* __restrict__ Wc2, const float* __restrict__ bc2,
    float* __restrict__ out)
{
  extern __shared__ _Float16 smem[];
  const int tid  = threadIdx.x;
  const int wave = tid >> 5;
  const int lane = tid & 31;
  const int c    = lane & 15;   // N column inside a 16x16 tile (C/D & B layouts)
  const int hi   = lane >> 4;   // half-select (M+8 / K+8 / K+16 per layouts)
  const int m16  = lane & 15;   // A-fragment row

  // ---- one-time: stage f16 transposed weights into LDS ----
  {
    const float* srcs[4] = {Wf1, Wf2, Wg1, Wg2};
    _Float16*    dsts[4] = {smem + OFF_WF1T, smem + OFF_WF2T,
                            smem + OFF_WG1T, smem + OFF_WG2T};
#pragma unroll
    for (int w = 0; w < 4; ++w) {
      const float* s = srcs[w];
      _Float16*    d = dsts[w];
      for (int i = tid; i < HID * HID; i += BLOCK) {
        int k = i >> 7, n = i & 127;        // source [K][N] -> dest [N][K]
        d[n * HID + k] = (_Float16)s[i];
      }
    }
    for (int i = tid; i < HID * 16; i += BLOCK) {
      int k = i >> 4, n = i & 15;           // Wc1 [128][16] -> [16][128]
      smem[OFF_WC1T + n * HID + k] = (_Float16)Wc1[i];
    }
    for (int i = tid; i < 16 * 32; i += BLOCK) {
      int n = i >> 5, k = i & 31;           // Wc2 [16][10] -> [16 N][32 K] padded
      float v = (n < NCLS && k < 16) ? Wc2[k * NCLS + n] : 0.0f;
      smem[OFF_WC2T + i] = (_Float16)v;
    }
  }
  __syncthreads();

  _Float16* stage = smem + OFF_STAGE + wave * (ROWS_PER_WAVE * HID);
  const int row0  = blockIdx.x * ROWS_PER_BLOCK + wave * ROWS_PER_WAVE;

  // per-lane bias scalars (N fixed per lane in C/D layout)
  float bf1v[8], bf2v[8], bg1v[8], bg2v[8];
#pragma unroll
  for (int nt = 0; nt < 8; ++nt) {
    bf1v[nt] = bf1[nt * 16 + c];
    bf2v[nt] = bf2[nt * 16 + c];
    bg1v[nt] = bg1[nt * 16 + c];
    bg2v[nt] = bg2[nt * 16 + c];
  }

  // ---- y0 = coeffs[:,0,:] @ Winit + binit (K=8, VALU; C/D layout in z) ----
  v8f z[8];
  {
    float wv[IN_DIM][8];
#pragma unroll
    for (int k = 0; k < IN_DIM; ++k)
#pragma unroll
      for (int nt = 0; nt < 8; ++nt)
        wv[k][nt] = Winit[k * HID + nt * 16 + c];
#pragma unroll
    for (int nt = 0; nt < 8; ++nt) {
      float b0 = binit[nt * 16 + c];
#pragma unroll
      for (int r = 0; r < 8; ++r) z[nt][r] = b0;
    }
#pragma unroll
    for (int r = 0; r < 8; ++r) {
      const float* cf = coeffs + (size_t)(row0 + r + 8 * hi) * (T_LEN * IN_DIM);
      float cfr[IN_DIM];
#pragma unroll
      for (int k = 0; k < IN_DIM; ++k) cfr[k] = cf[k];
#pragma unroll
      for (int nt = 0; nt < 8; ++nt) {
        float a = z[nt][r];
#pragma unroll
        for (int k = 0; k < IN_DIM; ++k) a = fmaf(cfr[k], wv[k][nt], a);
        z[nt][r] = a;
      }
    }
#pragma unroll
    for (int nt = 0; nt < 8; ++nt)
#pragma unroll
      for (int r = 0; r < 8; ++r)
        stage[(r + 8 * hi) * HID + nt * 16 + c] = (_Float16)z[nt][r];
  }

  // ---- 198 Euler-Maruyama steps, fully resident ----
  const float* dw_lane = dW + (size_t)(row0 + 8 * hi) * HID + c;

  // epilogue: tanh -> f16 staging store (temp array => independent TRANS ops,
  // no v_nop hazard stalls, and stores can be batched by the scheduler)
  auto epi_stage = [&](int nt, v8f acc) {
    float th[8];
#pragma unroll
    for (int r = 0; r < 8; ++r) th[r] = fast_tanh(acc[r]);
#pragma unroll
    for (int r = 0; r < 8; ++r)
      stage[(r + 8 * hi) * HID + nt * 16 + c] = (_Float16)th[r];
  };

#pragma unroll 1
  for (int s = 0; s < STEPS; ++s) {
    const float* dws = dw_lane + (size_t)s * (B_TOT * HID);

    // A(z) fragments (z also feeds GEMM3)
    v16h az[4];
#pragma unroll
    for (int kt = 0; kt < 4; ++kt)
      az[kt] = ld_frag(stage + m16 * HID + kt * 32 + hi * 8, 16);

    // GEMM1: h1 = tanh(z @ Wf1 + bf1) -> stage
    gemm_pass(smem + OFF_WF1T, az, bf1v, c, hi, epi_stage);

    v16h ah[4];
#pragma unroll
    for (int kt = 0; kt < 4; ++kt)
      ah[kt] = ld_frag(stage + m16 * HID + kt * 32 + hi * 8, 16);

    // GEMM2: z += tanh(h1 @ Wf2 + bf2) * dt
    gemm_pass(smem + OFF_WF2T, ah, bf2v, c, hi, [&](int nt, v8f acc) {
      float th[8];
#pragma unroll
      for (int r = 0; r < 8; ++r) th[r] = fast_tanh(acc[r]);
#pragma unroll
      for (int r = 0; r < 8; ++r) z[nt][r] = fmaf(th[r], DT_C, z[nt][r]);
    });

    // stream dW for this step straight into C/D layout; read-once -> NT.
    // Issued here: ~64 WMMAs (GEMM3+GEMM4) cover the latency, and the 64-VGPR
    // block is only live across the back half of the step.
    float dwv[8][8];
#pragma unroll
    for (int r = 0; r < 8; ++r)
#pragma unroll
      for (int nt = 0; nt < 8; ++nt)
        dwv[nt][r] = __builtin_nontemporal_load(dws + r * HID + nt * 16);

    if (s + 1 < STEPS) __builtin_prefetch(dws + (size_t)B_TOT * HID, 0, 0);

    // GEMM3: h2 = tanh(z @ Wg1 + bg1) -> stage (reuses az)
    gemm_pass(smem + OFF_WG1T, az, bg1v, c, hi, epi_stage);

#pragma unroll
    for (int kt = 0; kt < 4; ++kt)
      ah[kt] = ld_frag(stage + m16 * HID + kt * 32 + hi * 8, 16);

    // GEMM4: z += tanh(h2 @ Wg2 + bg2) * dW ; re-stage z as f16
    gemm_pass(smem + OFF_WG2T, ah, bg2v, c, hi, [&](int nt, v8f acc) {
      float th[8];
#pragma unroll
      for (int r = 0; r < 8; ++r) th[r] = fast_tanh(acc[r]);
#pragma unroll
      for (int r = 0; r < 8; ++r) {
        float zn = fmaf(th[r], dwv[nt][r], z[nt][r]);
        z[nt][r] = zn;
        stage[(r + 8 * hi) * HID + nt * 16 + c] = (_Float16)zn;
      }
    });
  }

  // ---- classifier: relu(z @ Wc1 + bc1) @ Wc2 + bc2 ----
  {
    v16h azf[4];
#pragma unroll
    for (int kt = 0; kt < 4; ++kt)
      azf[kt] = ld_frag(stage + m16 * HID + kt * 32 + hi * 8, 16);

    float b1 = bc1[c];
    v8f acc;
#pragma unroll
    for (int i = 0; i < 8; ++i) acc[i] = b1;
#pragma unroll
    for (int kt = 0; kt < 4; ++kt) {
      v16h b = ld_frag(smem + OFF_WC1T + c * HID + kt * 32 + hi * 8, 16);
      acc = wmma16(azf[kt], b, acc);
    }
    // relu -> zero-padded [16 M][32 K] f16 tile (reuse stage)
#pragma unroll
    for (int r = 0; r < 8; ++r) {
      stage[(r + 8 * hi) * 32 + c]      = (_Float16)fmaxf(acc[r], 0.0f);
      stage[(r + 8 * hi) * 32 + 16 + c] = (_Float16)0.0f;
    }
    v16h ahh = ld_frag(stage + m16 * 32 + hi * 8, 16);
    v16h bb  = ld_frag(smem + OFF_WC2T + c * 32 + hi * 8, 16);
    float b2 = (c < NCLS) ? bc2[c] : 0.0f;
    v8f o;
#pragma unroll
    for (int i = 0; i < 8; ++i) o[i] = b2;
    o = wmma16(ahh, bb, o);
    if (c < NCLS) {
#pragma unroll
      for (int r = 0; r < 8; ++r)
        out[(size_t)(row0 + r + 8 * hi) * NCLS + c] = o[r];
    }
  }
}

extern "C" void kernel_launch(void* const* d_in, const int* in_sizes, int n_in,
                              void* d_out, int out_size, void* d_ws, size_t ws_size,
                              hipStream_t stream) {
  (void)in_sizes; (void)n_in; (void)out_size; (void)d_ws; (void)ws_size;
  const float* coeffs = (const float*)d_in[0];
  // d_in[1] = times (unused: knots coincide with sample times)
  const float* dW    = (const float*)d_in[2];
  const float* Winit = (const float*)d_in[3];
  const float* binit = (const float*)d_in[4];
  const float* Wf1 = (const float*)d_in[5];
  const float* bf1 = (const float*)d_in[6];
  const float* Wf2 = (const float*)d_in[7];
  const float* bf2 = (const float*)d_in[8];
  const float* Wg1 = (const float*)d_in[9];
  const float* bg1 = (const float*)d_in[10];
  const float* Wg2 = (const float*)d_in[11];
  const float* bg2 = (const float*)d_in[12];
  const float* Wc1 = (const float*)d_in[13];
  const float* bc1 = (const float*)d_in[14];
  const float* Wc2 = (const float*)d_in[15];
  const float* bc2 = (const float*)d_in[16];
  float* out = (float*)d_out;

  dim3 grid(B_TOT / ROWS_PER_BLOCK);   // 64 blocks
  dim3 block(BLOCK);                   // 128 threads = 4 waves
  size_t smem_bytes = (size_t)SMEM_HALVES * sizeof(_Float16);  // ~149 KB dynamic LDS
  sde_fused_kernel<<<grid, block, smem_bytes, stream>>>(
      coeffs, dW, Winit, binit, Wf1, bf1, Wf2, bf2,
      Wg1, bg1, Wg2, bg2, Wc1, bc1, Wc2, bc2, out);
}
